// RNN_80135499808901
// MI455X (gfx1250) — matrix operand order repaired and checked by reading, not compiled, not run on hardware
//
#include <hip/hip_runtime.h>
#include <math.h>

// ---------------------------------------------------------------------------
// CDNA5 WMMA types / helpers (gfx1250, wave32)
// ---------------------------------------------------------------------------
typedef __attribute__((ext_vector_type(16))) __bf16 v16bf;
typedef __attribute__((ext_vector_type(8)))  float  v8f;

#define WMMA_BF16(a, b, c)                                                    \
  __builtin_amdgcn_wmma_f32_16x16x32_bf16(false, (a), false, (b), (short)0,   \
                                          (c), false, false)

// A-matrix 16x32 bf16 tile (ISA 7.12.2): lane l -> row m = l&15;
// elems 0..7  -> K = k0 + (l>>4)*8 + e ; elems 8..15 -> +16.
// Contiguous 8-element spans => lowers to two 16B loads (ds_load_b128 /
// global_load_b128) per tile.
static __device__ inline v16bf a_tile_bf16(const __bf16* row, int k0, int lane) {
  const int kb = k0 + ((lane >> 4) << 3);
  v16bf a;
#pragma unroll
  for (int e = 0; e < 8; ++e) a[e] = row[kb + e];
#pragma unroll
  for (int e = 0; e < 8; ++e) a[8 + e] = row[kb + 16 + e];
  return a;
}

static __device__ inline v16bf a_tile_f32(const float* row, int k0, int lane) {
  const int kb = k0 + ((lane >> 4) << 3);
  v16bf a;
#pragma unroll
  for (int e = 0; e < 8; ++e) a[e] = (__bf16)row[kb + e];
#pragma unroll
  for (int e = 0; e < 8; ++e) a[8 + e] = (__bf16)row[kb + 16 + e];
  return a;
}

// B-matrix 32x16 bf16 tile: lane l -> col n = l&15 (row of W, since B = W^T);
// elems 0..7 -> K = k0 + (l>>4)*8 + e ; elems 8..15 -> +16 (matches A).
static __device__ inline v16bf b_tile_bf16(const __bf16* wrow, int k0, int lane) {
  const int kb = k0 + ((lane >> 4) << 3);
  v16bf b;
#pragma unroll
  for (int e = 0; e < 8; ++e) b[e] = wrow[kb + e];
#pragma unroll
  for (int e = 0; e < 8; ++e) b[8 + e] = wrow[kb + 16 + e];
  return b;
}

static __device__ inline float fast_sigmoid(float x) {
  return 1.0f / (1.0f + __expf(-x));
}
static __device__ inline float fast_tanh(float x) {
  const float y = fminf(15.0f, fmaxf(-15.0f, x));
  const float e = __expf(-2.0f * y);
  return (1.0f - e) / (1.0f + e);
}

// ---------------------------------------------------------------------------
// fp32 -> bf16 convert (weights)
// ---------------------------------------------------------------------------
__global__ void k_cvt_bf16(const float* __restrict__ src,
                           __bf16* __restrict__ dst, int n) {
  int i = blockIdx.x * blockDim.x + threadIdx.x;
  if (i < n) dst[i] = (__bf16)src[i];
}

// cooperative global->LDS copy in 32-bit units (element count must be even)
static __device__ inline void stage_lds(const __bf16* __restrict__ g,
                                        __bf16* s, int nelem, int tid,
                                        int nthr) {
  const unsigned int* gs = (const unsigned int*)g;
  unsigned int* ds = (unsigned int*)s;
  const int n32 = nelem >> 1;
  for (int i = tid; i < n32; i += nthr) ds[i] = gs[i];
}

// ---------------------------------------------------------------------------
// rnn1: tanh RNN over the inner L=16 axis, one direction per blockIdx.y.
// Block = 256 thr = 8 waves; block owns 16 sequences; wave owns 16 hidden cols.
// acc = x_t * Wih^T + h * Whh^T ; h = tanh(acc + bih + bhh).
// Wih/Whh staged once into dynamic LDS (weight cache); h lives in one LDS buffer, rewritten each step.
// Dynamic smem layout: [ hbuf 16x128 | wih 128xKIN | whh 128x128 ] (bf16)
// ---------------------------------------------------------------------------
template <int KINBLK, bool XF32, bool OUTFULL>
__global__ void k_rnn1_recur(const void* __restrict__ Xin,
                             const __bf16* __restrict__ Wih,
                             const __bf16* __restrict__ Whh,
                             const float* __restrict__ bih,
                             const float* __restrict__ bhh,
                             __bf16* __restrict__ out) {
  const int d    = blockIdx.y;                 // direction
  const int seq0 = blockIdx.x << 4;            // 16 sequences per block
  const int lane = threadIdx.x & 31;
  const int wv   = threadIdx.x >> 5;           // 0..7 : hidden tile
  constexpr int KIN = KINBLK * 32;

  extern __shared__ __bf16 smem[];
  __bf16* hbuf  = smem;                        // 16*128
  __bf16* wih_s = smem + 16 * 128;             // 128*KIN
  __bf16* whh_s = wih_s + 128 * KIN;           // 128*128

  for (int i = threadIdx.x; i < 16 * 128; i += blockDim.x)
    hbuf[i] = (__bf16)0.0f;
  stage_lds(Wih + (size_t)d * 128 * KIN, wih_s, 128 * KIN, threadIdx.x,
            blockDim.x);
  stage_lds(Whh + (size_t)d * 128 * 128, whh_s, 128 * 128, threadIdx.x,
            blockDim.x);

  const int ncol = (wv << 4) + (lane & 15);
  const float bias = bih[d * 128 + ncol] + bhh[d * 128 + ncol];
  const __bf16* wihrow = wih_s + (size_t)ncol * KIN;
  const __bf16* whhrow = whh_s + (size_t)ncol * 128;
  __syncthreads();

  const int m = lane & 15;
  const __bf16* hrow = hbuf + m * 128;

  for (int s = 0; s < 16; ++s) {
    const int t = d ? (15 - s) : s;
    v8f acc = {};
    if (XF32) {
      const float* xr = (const float*)Xin + ((size_t)(seq0 + m) * 16 + t) * KIN;
#pragma unroll
      for (int kb = 0; kb < KINBLK; ++kb)
        acc = WMMA_BF16(a_tile_f32(xr, kb << 5, lane),
                        b_tile_bf16(wihrow, kb << 5, lane), acc);
    } else {
      const __bf16* xr =
          (const __bf16*)Xin + ((size_t)(seq0 + m) * 16 + t) * KIN;
#pragma unroll
      for (int kb = 0; kb < KINBLK; ++kb)
        acc = WMMA_BF16(a_tile_bf16(xr, kb << 5, lane),
                        b_tile_bf16(wihrow, kb << 5, lane), acc);
    }
#pragma unroll
    for (int kb = 0; kb < 4; ++kb)
      acc = WMMA_BF16(a_tile_bf16(hrow, kb << 5, lane),
                      b_tile_bf16(whhrow, kb << 5, lane), acc);

    __syncthreads();  // all waves done reading h before it is overwritten
#pragma unroll
    for (int r = 0; r < 8; ++r) {
      const int mr = r + ((lane >> 4) << 3);   // D-layout row
      const float v = fast_tanh(acc[r] + bias);
      hbuf[mr * 128 + ncol] = (__bf16)v;
      if (OUTFULL) {
        out[((size_t)(seq0 + mr) * 16 + t) * 256 + (d << 7) + ncol] = (__bf16)v;
      } else if (t == 15) {
        out[(size_t)(seq0 + mr) * 256 + (d << 7) + ncol] = (__bf16)v;
      }
    }
    __syncthreads();
  }
}

// ---------------------------------------------------------------------------
// GRU recurrence over T=512. One block per direction, 512 thr = 16 waves.
// wave -> (m-tile of 16 batch rows, 16-wide j column tile); each wave computes
// all three gate tiles (r,z,n) for its columns so the h-update is wave-local.
// Whh_d (384x128 bf16, 96 KB) staged in dynamic LDS; B tiles re-read per step
// via ds_load_b128 (no VGPR spill, unlike register-resident weights).
// Dynamic smem layout: [ hbuf 32x128 | whh 384x128 ] (bf16)
// ---------------------------------------------------------------------------
__global__ void k_gru_recur(const float* __restrict__ gxw,
                            const __bf16* __restrict__ Whh,
                            const float* __restrict__ bhh,
                            __bf16* __restrict__ go) {
  const int d    = blockIdx.x;
  const int lane = threadIdx.x & 31;
  const int wv   = threadIdx.x >> 5;  // 0..15
  const int mt   = wv >> 3;           // 0..1 : batch tile (B=32)
  const int jt   = wv & 7;            // 0..7 : hidden column tile
  const int jcol = (jt << 4) + (lane & 15);

  extern __shared__ __bf16 smem[];
  __bf16* hbuf  = smem;               // 32*128
  __bf16* whh_s = smem + 32 * 128;    // 384*128

  for (int i = threadIdx.x; i < 32 * 128; i += blockDim.x)
    hbuf[i] = (__bf16)0.0f;
  stage_lds(Whh + (size_t)d * 384 * 128, whh_s, 384 * 128, threadIdx.x,
            blockDim.x);

  float bh[3];
  const __bf16* wrow[3];
#pragma unroll
  for (int g = 0; g < 3; ++g) {
    wrow[g] = whh_s + (size_t)((g << 7) + jcol) * 128;
    bh[g] = bhh[d * 384 + (g << 7) + jcol];
  }
  __syncthreads();

  const int m = (mt << 4) + (lane & 15);
  const __bf16* hrow = hbuf + m * 128;

  for (int s = 0; s < 512; ++s) {
    const int t = d ? (511 - s) : s;
    v8f acc[3] = {};
#pragma unroll
    for (int kb = 0; kb < 4; ++kb) {
      v16bf a = a_tile_bf16(hrow, kb << 5, lane);
#pragma unroll
      for (int g = 0; g < 3; ++g)
        acc[g] = WMMA_BF16(a, b_tile_bf16(wrow[g], kb << 5, lane), acc[g]);
    }
    __syncthreads();
#pragma unroll
    for (int r = 0; r < 8; ++r) {
      const int mr = (mt << 4) + r + ((lane >> 4) << 3);  // batch row 0..31
      const size_t row = (size_t)mr * 512 + t;            // b*T + t
      const float* xp = gxw + row * 768 + d * 384;
      const float xr = xp[jcol];
      const float xz = xp[128 + jcol];
      const float xn = xp[256 + jcol];
      const float rg = fast_sigmoid(xr + acc[0][r] + bh[0]);
      const float zg = fast_sigmoid(xz + acc[1][r] + bh[1]);
      const float gn = acc[2][r] + bh[2];
      const float hold = (float)hbuf[mr * 128 + jcol];   // element owned by me
      const float ng = fast_tanh(xn + rg * gn);
      const float hn = (1.0f - zg) * ng + zg * hold;
      hbuf[mr * 128 + jcol] = (__bf16)hn;
      go[row * 256 + (d << 7) + jcol] = (__bf16)hn;
    }
    __syncthreads();
  }
}

// ---------------------------------------------------------------------------
// Generic WMMA GEMM: C(M,N) = A(M,K)bf16 * W(N,K)bf16^T + bias.
// If `lengths` != null: ragged scatter (FC layer) into `scat` with N cols,
// T=512 rows per sample; else dense fp32 store into C.
// ---------------------------------------------------------------------------
__global__ void k_gemm_bf16nt(const __bf16* __restrict__ A,
                              const __bf16* __restrict__ W,
                              const float* __restrict__ bias,
                              float* __restrict__ C, int M, int N, int K,
                              const int* __restrict__ lengths,
                              float* __restrict__ scat) {
  const int lane   = threadIdx.x & 31;
  const int wv     = threadIdx.x >> 5;
  const int ntile  = blockIdx.y * 8 + wv;
  const int ntiles = (N + 15) >> 4;
  if (ntile >= ntiles) return;  // wave-uniform

  const int m0 = blockIdx.x << 4;
  const int n  = (ntile << 4) + (lane & 15);
  const __bf16* arow = A + (size_t)(m0 + (lane & 15)) * K;
  const __bf16* wrow = W + (size_t)n * K;
  const int kblk = K >> 5;

  v8f acc = {};
  for (int kb = 0; kb < kblk; ++kb) {
    v16bf a = a_tile_bf16(arow, kb << 5, lane);
    v16bf b;
    if (n < N) {
      b = b_tile_bf16(wrow, kb << 5, lane);
    } else {
#pragma unroll
      for (int e = 0; e < 16; ++e) b[e] = (__bf16)0.0f;
    }
    acc = WMMA_BF16(a, b, acc);
  }
  const float bn = (n < N) ? bias[n] : 0.0f;

  if (lengths) {
    const int bsmp  = m0 >> 9;  // T = 512, blocks never straddle samples
    int offr = 0;
    for (int i = 0; i < bsmp; ++i) offr += lengths[i];
    const int len   = lengths[bsmp];
    const int tbase = m0 & 511;
#pragma unroll
    for (int r = 0; r < 8; ++r) {
      const int t = tbase + r + ((lane >> 4) << 3);
      if (t < len && n < N)
        scat[(size_t)(offr + t) * N + n] = acc[r] + bn;
    }
  } else {
#pragma unroll
    for (int r = 0; r < 8; ++r) {
      const int mr = m0 + r + ((lane >> 4) << 3);
      if (n < N) C[(size_t)mr * N + n] = acc[r] + bn;
    }
  }
}

// ---------------------------------------------------------------------------
// Host orchestration
// ---------------------------------------------------------------------------
extern "C" void kernel_launch(void* const* d_in, const int* in_sizes, int n_in,
                              void* d_out, int out_size, void* d_ws,
                              size_t ws_size, hipStream_t stream) {
  (void)in_sizes; (void)n_in; (void)out_size; (void)ws_size;

  const float* x        = (const float*)d_in[0];
  const int*   lengths  = (const int*)  d_in[1];
  const float* r1l0_Wih = (const float*)d_in[2];
  const float* r1l0_Whh = (const float*)d_in[3];
  const float* r1l0_bih = (const float*)d_in[4];
  const float* r1l0_bhh = (const float*)d_in[5];
  const float* r1l1_Wih = (const float*)d_in[6];
  const float* r1l1_Whh = (const float*)d_in[7];
  const float* r1l1_bih = (const float*)d_in[8];
  const float* r1l1_bhh = (const float*)d_in[9];
  const float* g0_Wih   = (const float*)d_in[10];
  const float* g0_Whh   = (const float*)d_in[11];
  const float* g0_bih   = (const float*)d_in[12];
  const float* g0_bhh   = (const float*)d_in[13];
  const float* g1_Wih   = (const float*)d_in[14];
  const float* g1_Whh   = (const float*)d_in[15];
  const float* g1_bih   = (const float*)d_in[16];
  const float* g1_bhh   = (const float*)d_in[17];
  const float* fc_W     = (const float*)d_in[18];
  const float* fc_b     = (const float*)d_in[19];

  // deterministic bump allocator over d_ws
  char* ws = (char*)d_ws;
  size_t off = 0;
  auto alloc = [&](size_t bytes) -> void* {
    void* p = ws + off;
    off += (bytes + 255) & ~(size_t)255;
    return p;
  };

  const size_t BT = 16384;  // B*T

  __bf16* h0out = (__bf16*)alloc(BT * 16 * 256 * 2);  // rnn1 L0 full seq
  __bf16* last1 = (__bf16*)alloc(BT * 256 * 2);       // rnn1 L1 last step
  float*  gxw   = (float*) alloc(BT * 768 * 4);       // GRU gate proj (reused)
  __bf16* go0   = (__bf16*)alloc(BT * 256 * 2);       // GRU L0 output
  __bf16* go1   = (__bf16*)alloc(BT * 256 * 2);       // GRU L1 output

  __bf16* wb_r1l0_Wih = (__bf16*)alloc((size_t)32768  * 2);
  __bf16* wb_r1l0_Whh = (__bf16*)alloc((size_t)32768  * 2);
  __bf16* wb_r1l1_Wih = (__bf16*)alloc((size_t)65536  * 2);
  __bf16* wb_r1l1_Whh = (__bf16*)alloc((size_t)32768  * 2);
  __bf16* wb_g0_Wih   = (__bf16*)alloc((size_t)196608 * 2);
  __bf16* wb_g0_Whh   = (__bf16*)alloc((size_t)98304  * 2);
  __bf16* wb_g1_Wih   = (__bf16*)alloc((size_t)196608 * 2);
  __bf16* wb_g1_Whh   = (__bf16*)alloc((size_t)98304  * 2);
  __bf16* wb_fcW      = (__bf16*)alloc((size_t)15616  * 2);

  auto cvt = [&](const float* s, __bf16* d, int n) {
    k_cvt_bf16<<<(n + 255) / 256, 256, 0, stream>>>(s, d, n);
  };
  cvt(r1l0_Wih, wb_r1l0_Wih, 32768);
  cvt(r1l0_Whh, wb_r1l0_Whh, 32768);
  cvt(r1l1_Wih, wb_r1l1_Wih, 65536);
  cvt(r1l1_Whh, wb_r1l1_Whh, 32768);
  cvt(g0_Wih,   wb_g0_Wih,   196608);
  cvt(g0_Whh,   wb_g0_Whh,   98304);
  cvt(g1_Wih,   wb_g1_Wih,   196608);
  cvt(g1_Whh,   wb_g1_Whh,   98304);
  cvt(fc_W,     wb_fcW,      15616);

  // dynamic LDS sizes (bf16 elements * 2 bytes)
  const size_t shm_l0  = (size_t)(16 * 128 + 128 * 128 + 128 * 128) * 2; // 68KB
  const size_t shm_l1  = (size_t)(16 * 128 + 128 * 256 + 128 * 128) * 2; // 100KB
  const size_t shm_gru = (size_t)(32 * 128 + 384 * 128) * 2;             // 104KB

  // rnn1 layer0: x(fp32,K=128) -> h0out (full inner sequence, bf16, 2H concat)
  k_rnn1_recur<4, true, true><<<dim3(1024, 2), 256, shm_l0, stream>>>(
      (const void*)x, wb_r1l0_Wih, wb_r1l0_Whh, r1l0_bih, r1l0_bhh, h0out);

  // rnn1 layer1: h0out(bf16,K=256) -> last inner timestep only
  k_rnn1_recur<8, false, false><<<dim3(1024, 2), 256, shm_l1, stream>>>(
      (const void*)h0out, wb_r1l1_Wih, wb_r1l1_Whh, r1l1_bih, r1l1_bhh, last1);

  // GRU layer0: input projection then recurrence
  k_gemm_bf16nt<<<dim3(1024, 6), 256, 0, stream>>>(
      last1, wb_g0_Wih, g0_bih, gxw, (int)BT, 768, 256, nullptr, nullptr);
  k_gru_recur<<<2, 512, shm_gru, stream>>>(gxw, wb_g0_Whh, g0_bhh, go0);

  // GRU layer1 (gxw buffer reused)
  k_gemm_bf16nt<<<dim3(1024, 6), 256, 0, stream>>>(
      go0, wb_g1_Wih, g1_bih, gxw, (int)BT, 768, 256, nullptr, nullptr);
  k_gru_recur<<<2, 512, shm_gru, stream>>>(gxw, wb_g1_Whh, g1_bhh, go1);

  // FC + fused ragged scatter into d_out (C = 61)
  k_gemm_bf16nt<<<dim3(1024, 1), 256, 0, stream>>>(
      go1, wb_fcW, fc_b, nullptr, (int)BT, 61, 256, lengths, (float*)d_out);
}
